// NeuralSheet_8632884265060
// MI455X (gfx1250) — matrix operand order repaired and checked by reading, not compiled, not run on hardware
//
#include <hip/hip_runtime.h>

// ---------------------------------------------------------------------------
// NeuralSheet on MI455X (gfx1250):
//   W (4096x4096) precomputed to bf16 in workspace (32MB, L2-resident on the
//   192MB L2).  20 sequential iterations -> 19 WMMA matvec kernel launches
//   (first iteration folded into the afferent kernel since r0 = 0).
//   r / x vectors staged into LDS with the Tensor Data Mover.
// ---------------------------------------------------------------------------

#define SN 4096        // S*S  (and I*I)
#define KC 1024        // K-range per wave (4-way K split)

typedef __attribute__((ext_vector_type(16))) __bf16 v16bf;
typedef __attribute__((ext_vector_type(8)))  float  v8f;
typedef __attribute__((ext_vector_type(4)))  unsigned int u32x4;
typedef __attribute__((ext_vector_type(8)))  int          i32x8;
typedef __attribute__((ext_vector_type(4)))  int          i32x4;

union BV { uint4 u[2]; v16bf v; };

// ---------------------------------------------------------------------------
// Tensor Data Mover: 1-D tile copy global -> LDS.
// D# built per CDNA5 ISA ch.10.8: 8-byte elements, tile_dim0 = tensor_dim0 =
// nbytes/8, stride = dim0.  Issue from one wave; completion via TENSORcnt.
// ---------------------------------------------------------------------------
__device__ __forceinline__ void tdm_load_1d(const void* gsrc, unsigned lds_off,
                                            unsigned nbytes)
{
    const unsigned n8 = nbytes >> 3;                    // 8-byte elements
    const unsigned long long ga = (unsigned long long)gsrc;

    u32x4 g0;
    g0[0] = 1u;                                         // count=1, user desc
    g0[1] = lds_off;                                    // lds_addr
    g0[2] = (unsigned)(ga & 0xFFFFFFFFu);               // global_addr[31:0]
    g0[3] = (unsigned)((ga >> 32) & 0x01FFFFFFu)        // global_addr[56:32]
          | (2u << 30);                                 // type = 2 ("image")

    i32x8 g1;
    g1[0] = (int)(3u << 16);                            // mask=0, data_size=3 (8B)
    g1[1] = (int)((n8 & 0xFFFFu) << 16);                // tensor_dim0[15:0]
    g1[2] = (int)(((n8 >> 16) & 0xFFFFu) | (1u << 16)); // dim0[31:16], dim1=1
    g1[3] = (int)((n8 & 0xFFFFu) << 16);                // tile_dim0
    g1[4] = 0;                                          // tile_dim1/2 unused
    g1[5] = (int)n8;                                    // tensor_dim0_stride lo
    g1[6] = 0;
    g1[7] = 0;

    i32x4 z4 = {0, 0, 0, 0};
#if __clang_major__ >= 23
    i32x8 z8 = {0, 0, 0, 0, 0, 0, 0, 0};
    __builtin_amdgcn_tensor_load_to_lds(g0, g1, z4, z4, z8, 0);
#else
    __builtin_amdgcn_tensor_load_to_lds(g0, g1, z4, z4, 0);
#endif
}

// ---------------------------------------------------------------------------
// Kernel 1: W_bf16[i][j] = bf16( exc[i][j] - (corr*mask)[i][j] / (rowsum+eps) )
// One block per row; products staged in LDS so corr/mask are read once.
// ---------------------------------------------------------------------------
__global__ __launch_bounds__(256) void prep_w_kernel(
    const float* __restrict__ exc, const float* __restrict__ corr,
    const float* __restrict__ mask, __bf16* __restrict__ Wb)
{
    __shared__ float p_lds[SN];     // 16 KB
    __shared__ float red[256];

    const int i = blockIdx.x, tid = threadIdx.x;
    const float* cr = corr + (size_t)i * SN;
    const float* mk = mask + (size_t)i * SN;
    const float* ex = exc  + (size_t)i * SN;

    float local = 0.0f;
#pragma unroll
    for (int t = 0; t < 16; ++t) {
        int j = tid + t * 256;
        float p = cr[j] * mk[j];
        p_lds[j] = p;
        local += p;
    }
    red[tid] = local;
    __syncthreads();
    for (int s = 128; s > 0; s >>= 1) {
        if (tid < s) red[tid] += red[tid + s];
        __syncthreads();
    }
    const float inv = 1.0f / (red[0] + 1e-11f);

    __bf16* wb = Wb + (size_t)i * SN;
#pragma unroll
    for (int t = 0; t < 16; ++t) {
        int j = tid + t * 256;
        wb[j] = (__bf16)(ex[j] - p_lds[j] * inv);
    }
}

// ---------------------------------------------------------------------------
// Kernel 2: base[i] = dot(A[i,:], x) - thr[i];  r1 = tanh(relu(base))
// (iteration 1 of the recurrence, since r0 == 0).  x staged via TDM.
// ---------------------------------------------------------------------------
__global__ __launch_bounds__(256) void afferent_kernel(
    const float* __restrict__ A, const float* __restrict__ x,
    const float* __restrict__ thr, float* __restrict__ base,
    __bf16* __restrict__ r0_bf16, float* __restrict__ out)
{
    __shared__ __align__(16) float xs[SN];   // 16 KB
    __shared__ float red[256];

    const int i = blockIdx.x, tid = threadIdx.x;
    if (tid < 32) {                           // wave 0 drives the TDM
        tdm_load_1d(x, (unsigned)(size_t)(void*)xs, SN * 4);
        __builtin_amdgcn_s_wait_tensorcnt(0);
    }
    __syncthreads();

    const float* Ar = A + (size_t)i * SN;
    float local = 0.0f;
#pragma unroll
    for (int t = 0; t < 16; ++t) {
        int j = tid + t * 256;
        local += Ar[j] * xs[j];
    }
    red[tid] = local;
    __syncthreads();
    for (int s = 128; s > 0; s >>= 1) {
        if (tid < s) red[tid] += red[tid + s];
        __syncthreads();
    }
    if (tid == 0) {
        float b = red[0] - thr[i];
        base[i] = b;
        float z = b > 0.0f ? b : 0.0f;
        float r = tanhf(z);
        out[i]     = r;
        r0_bf16[i] = (__bf16)r;
    }
}

// ---------------------------------------------------------------------------
// Kernel 3: one recurrence iteration:  r_dst = tanh(relu(base + 2*(W @ r_src)))
// Block = 256 threads (8 waves), owns 32 output rows = 2 tiles of 16.
// Per tile, 4 waves split K (1024 each); each wave runs
// V_WMMA_F32_16X16X32_BF16 with A = W-tile, B = r-chunk in column N=0 only.
// Lanes with N!=0 read a zero pad in LDS (stride 0) -> no per-iteration
// masking VALU and no WMMA->VALU hazard NOPs in the hot loop.
// Grid = 4096/32 = 128 blocks.  W streams from L2 (32MB bf16).
// ---------------------------------------------------------------------------
__global__ __launch_bounds__(256) void iter_kernel(
    const __bf16* __restrict__ Wb, const float* __restrict__ base,
    const __bf16* __restrict__ rsrc, __bf16* __restrict__ rdst,
    float* __restrict__ out)
{
    __shared__ __align__(16) __bf16 r_lds[SN];   // 8 KB: whole r vector, bf16
    __shared__ __align__(16) __bf16 zpad[16];    // 32 B of zeros for N!=0 lanes
    __shared__ float part[8][16];                // per-wave column-0 partials

    const int tid = threadIdx.x;
    if (tid < 8) ((unsigned*)zpad)[tid] = 0u;
    if (tid < 32) {                              // wave 0 DMAs r into LDS
        tdm_load_1d(rsrc, (unsigned)(size_t)(void*)r_lds, SN * 2);
        __builtin_amdgcn_s_wait_tensorcnt(0);
    }
    __syncthreads();

    const int lane = tid & 31;
    const int w    = tid >> 5;         // wave 0..7
    const int tile = w >> 2;           // 0..1   -> which 16-row tile
    const int kc   = w & 3;            // 0..3   -> which K quarter
    const int m0   = blockIdx.x * 32 + tile * 16;
    const int row  = m0 + (lane & 15);
    const int hi   = (lane & 16) ? 1 : 0;
    const bool keep = ((lane & 15) == 0);        // N==0 column carries r

    // A as loaded here (contiguous k order, 16x32): lanes 0-15 hold K0..15 of row M=lane,
    // lanes 16-31 hold K16..31 of row M=lane-16.
    const __bf16* aptr = Wb + (size_t)row * SN + (size_t)(kc * KC + hi * 16);
    // B in the same contiguous k order (32x16): lanes 0-15 hold K0..15 (N=lane),
    // lanes 16-31 hold K16..31 (N=lane-16).  Only N==0 lanes advance through
    // r; all other lanes re-read the 32-byte zero pad (stride 0).
    const char* bp   = (const char*)(keep ? (r_lds + kc * KC + hi * 16) : zpad);
    const int  bstep = keep ? 64 : 0;            // bytes per K-step of 32

    v8f acc = {};
#pragma unroll 4
    for (int k = 0; k < KC; k += 32) {
        BV a, b;
        a.u[0] = *(const uint4*)(aptr);          // K +0..7   (or +16..23)
        a.u[1] = *(const uint4*)(aptr + 8);      // K +8..15  (or +24..31)
        aptr += 32;
        b.u[0] = *(const uint4*)(bp);
        b.u[1] = *(const uint4*)(bp + 16);
        bp += bstep;
        acc = __builtin_amdgcn_wmma_f32_16x16x32_bf16(
                  false, a.v, false, b.v, (short)0, acc, false, false);
    }

    // Column N=0 of D: lane 0 -> rows m0+0..7 (VGPR 0..7),
    //                  lane 16 -> rows m0+8..15.
    if (keep) {
        const int half = hi * 8;
#pragma unroll
        for (int v = 0; v < 8; ++v) part[w][half + v] = acc[v];
    }
    __syncthreads();

    if (tid < 32) {
        const int t2   = tid >> 4;     // tile
        const int rloc = tid & 15;
        const float y = part[t2 * 4 + 0][rloc] + part[t2 * 4 + 1][rloc]
                      + part[t2 * 4 + 2][rloc] + part[t2 * 4 + 3][rloc];
        const int grow = blockIdx.x * 32 + tid;
        float z = base[grow] + 2.0f * y;           // LATERAL_STRENGTH = 2
        z = z > 0.0f ? z : 0.0f;                   // relu
        const float r = tanhf(z);
        out[grow]  = r;
        rdst[grow] = (__bf16)r;
    }
}

// ---------------------------------------------------------------------------
extern "C" void kernel_launch(void* const* d_in, const int* in_sizes, int n_in,
                              void* d_out, int out_size, void* d_ws, size_t ws_size,
                              hipStream_t stream)
{
    const float* x    = (const float*)d_in[0];   // input_crop        (4096)
    const float* aw   = (const float*)d_in[1];   // afferent_weights  (4096*4096)
    const float* exc  = (const float*)d_in[2];   // lateral_weights_exc
    const float* corr = (const float*)d_in[3];   // lateral_correlations
    const float* mask = (const float*)d_in[4];   // masks
    const float* thr  = (const float*)d_in[5];   // thresholds        (4096)
    float* out = (float*)d_out;

    char* ws = (char*)d_ws;
    const size_t W_BYTES = (size_t)SN * SN * 2;            // 32 MB bf16 W
    __bf16* Wb   = (__bf16*)ws;
    float*  base = (float*)(ws + W_BYTES);                 // 16 KB
    __bf16* rb0  = (__bf16*)(ws + W_BYTES + 16384);        // 8 KB ping
    __bf16* rb1  = (__bf16*)(ws + W_BYTES + 16384 + 8192); // 8 KB pong

    // Build W (bf16) and afferent base; afferent kernel also performs
    // iteration 1 of the recurrence (r0 == 0 -> r1 = tanh(relu(base))).
    prep_w_kernel<<<SN, 256, 0, stream>>>(exc, corr, mask, Wb);
    afferent_kernel<<<SN, 256, 0, stream>>>(aw, x, thr, base, rb0, out);

    // Iterations 2..20: ping-pong r between launches (grid-wide dependency).
    __bf16* src = rb0;
    __bf16* dst = rb1;
    for (int it = 1; it < 20; ++it) {
        iter_kernel<<<SN / 32, 256, 0, stream>>>(Wb, base, src, dst, out);
        __bf16* t = src; src = dst; dst = t;
    }
}
